// DynaLoRALinear_46729244180518
// MI455X (gfx1250) — compile-verified, matrix operand order checked by reading, not run on hardware
//
#include <hip/hip_runtime.h>
#include <stdint.h>

#define DEV __device__ __forceinline__

typedef __attribute__((ext_vector_type(16))) __bf16          bf16x16;
typedef __attribute__((ext_vector_type(16))) unsigned short  ushort16;
typedef __attribute__((ext_vector_type(8)))  unsigned short  ushort8;
typedef __attribute__((ext_vector_type(8)))  float           floatx8;

constexpr int kBat = 4, kL = 4096, kD = 1024, kE = 8, kR = 8, kDH = 512;
constexpr int kTok = kBat * kL;   // 16384 tokens
constexpr int kER  = kE * kR;     // 64 (flattened expert*rank)

constexpr int BM = 128, BK = 32, PAD = 8, LDSLD = BK + PAD;  // 40 shorts = 80 B row stride

DEV unsigned short f32_to_bf16(float f) {
    union { float f; uint32_t u; } c{f};
    uint32_t u = c.u;
    u += 0x7FFFu + ((u >> 16) & 1u);            // round-to-nearest-even
    return (unsigned short)(u >> 16);
}
DEV float bf16_to_f32(unsigned short h) {
    union { uint32_t u; float f; } c{((uint32_t)h) << 16};
    return c.f;
}

// ---------------------------------------------------------------------------
// Prep: convert x + all contraction weights to bf16; repack lora_B -> U[d][er]
// ---------------------------------------------------------------------------
__global__ __launch_bounds__(256) void prep_kernel(
    const float* __restrict__ x, const float* __restrict__ Wb,
    const float* __restrict__ W1, const float* __restrict__ lA,
    const float* __restrict__ lB,
    unsigned short* __restrict__ xb, unsigned short* __restrict__ wbb,
    unsigned short* __restrict__ w1b, unsigned short* __restrict__ a2b,
    unsigned short* __restrict__ ub) {
    const long i = (long)blockIdx.x * blockDim.x + threadIdx.x;
    const long stride = (long)gridDim.x * blockDim.x;
    const long n_x = (long)kTok * kD;
    const long n_wb = (long)kD * kD, n_w1 = (long)kDH * kD;
    const long n_a = (long)kER * kD, n_u = (long)kD * kER;
    for (long t = i; t < n_x;  t += stride) xb[t]  = f32_to_bf16(x[t]);
    for (long t = i; t < n_wb; t += stride) wbb[t] = f32_to_bf16(Wb[t]);
    for (long t = i; t < n_w1; t += stride) w1b[t] = f32_to_bf16(W1[t]);
    for (long t = i; t < n_a;  t += stride) a2b[t] = f32_to_bf16(lA[t]);  // [E,R,D] already row-major over D
    for (long t = i; t < n_u;  t += stride) {                             // U[d][e*8+r] = lora_B[e][d][r]
        const int d = (int)(t >> 6), er = (int)(t & 63);
        const int e = er >> 3, r = er & 7;
        ub[t] = f32_to_bf16(lB[(long)e * kD * kR + (long)d * kR + r]);
    }
}

// ---------------------------------------------------------------------------
// Async global -> LDS (CDNA5 path, tracked by ASYNCcnt). 16 B per lane.
// ---------------------------------------------------------------------------
DEV void async_chunk(const unsigned short* __restrict__ sd,
                     const unsigned short* __restrict__ g) {
    const uint32_t lds_off = (uint32_t)(uintptr_t)sd;   // addr[31:0] = LDS byte offset
    asm volatile("global_load_async_to_lds_b128 %0, %1, off"
                 :: "v"(lds_off), "v"(g)
                 : "memory");
}
DEV void wait_async0() { asm volatile("s_wait_asynccnt 0x0" ::: "memory"); }

// Issue one [rows x BK] tile copy, row-major source with leading dim ldg.
DEV void issue_tile(unsigned short* __restrict__ sd, int rows,
                    const unsigned short* __restrict__ g,
                    long row0, int k0, int ldg, int tid) {
    constexpr int CPR = BK / 8;                 // 16-byte chunks per row (4)
    const int nch = rows * CPR;
    for (int c = tid; c < nch; c += 256) {
        const int r  = c / CPR;
        const int kc = (c % CPR) << 3;
        async_chunk(sd + r * LDSLD + kc,
                    g + (row0 + r) * (long)ldg + k0 + kc);
    }
}

// A/B fragment per ISA layout: lanes 0-15 K=[0..7,16..23], lanes 16-31 K=[8..15,24..31]
DEV bf16x16 frag_lds(const unsigned short* __restrict__ sd, int row, int koff) {
    const unsigned short* p = sd + row * LDSLD + koff;
    ushort8 lo = *(const ushort8*)p;
    ushort8 hi = *(const ushort8*)(p + 16);
    ushort16 u;
#pragma unroll
    for (int i = 0; i < 8; ++i) { u[i] = lo[i]; u[i + 8] = hi[i]; }
    return __builtin_bit_cast(bf16x16, u);
}

// MODE 0: h = relu(acc + bias), store bf16      (aux = b1[N])
// MODE 1: y = acc * gates[row][col>>3], bf16    (aux = gates[M x 8])
// MODE 2: out = acc + fused (Y[M,64] * U[N,64]^T), store f32
template <int MODE, int BN>
__global__ __launch_bounds__(256) void gemm_bf16_kernel(
    const unsigned short* __restrict__ A, const unsigned short* __restrict__ Bw,
    int K, const float* __restrict__ aux,
    const unsigned short* __restrict__ Y, const unsigned short* __restrict__ U,
    unsigned short* __restrict__ out16, float* __restrict__ out32, int ldn) {
    constexpr int WAVE_N = BN / 64;
    constexpr int WAVE_M = 8 / WAVE_N;
    constexpr int TM = BM / WAVE_M;
    constexpr int MI = TM / 16;
    constexpr int NI = 4;

    __shared__ unsigned short As[2][BM * LDSLD];   // double-buffered
    __shared__ unsigned short Bs[2][BN * LDSLD];

    const int tid  = threadIdx.x;
    const int wave = tid >> 5;
    const int lane = tid & 31;
    const int wm = wave % WAVE_M;
    const int wn = wave / WAVE_M;
    const int lrow = lane & 15;
    const int koff = (lane >> 4) << 3;   // K sub-offset for A/B fragments
    const int m0 = blockIdx.x * BM;
    const int n0 = blockIdx.y * BN;

    floatx8 acc[MI][NI] = {};

    auto mma_step = [&](const unsigned short* asd, const unsigned short* bsd) {
        bf16x16 bfrag[NI];
#pragma unroll
        for (int ni = 0; ni < NI; ++ni)
            bfrag[ni] = frag_lds(bsd, wn * 64 + ni * 16 + lrow, koff);
#pragma unroll
        for (int mi = 0; mi < MI; ++mi) {
            bf16x16 afrag = frag_lds(asd, wm * TM + mi * 16 + lrow, koff);
#pragma unroll
            for (int ni = 0; ni < NI; ++ni)
                acc[mi][ni] = __builtin_amdgcn_wmma_f32_16x16x32_bf16(
                    false, afrag, false, bfrag[ni], (short)0, acc[mi][ni],
                    false, false);
        }
    };

    const int nsteps = K / BK;
    const int total  = nsteps + (MODE == 2 ? kER / BK : 0);  // + LoRA-up steps

    auto issue_step = [&](int s, int buf) {
        if (s < nsteps) {
            issue_tile(As[buf], BM, A,  m0, s * BK, K, tid);
            issue_tile(Bs[buf], BN, Bw, n0, s * BK, K, tid);
        } else if (MODE == 2) {
            issue_tile(As[buf], BM, Y, m0, (s - nsteps) * BK, kER, tid);
            issue_tile(Bs[buf], BN, U, n0, (s - nsteps) * BK, kER, tid);
        }
    };

    issue_step(0, 0);
    for (int s = 0; s < total; ++s) {
        wait_async0();          // my tile-s chunks landed in LDS
        __syncthreads();        // everyone's landed; buf s^1 free to overwrite
        if (s + 1 < total) issue_step(s + 1, (s + 1) & 1);   // prefetch next tile
        mma_step(As[s & 1], Bs[s & 1]);                      // overlap with async DMA
    }

    const int rhalf = (lane >> 4) << 3;  // C/D layout: lanes 16-31 hold M=8..15
#pragma unroll
    for (int mi = 0; mi < MI; ++mi) {
#pragma unroll
        for (int ni = 0; ni < NI; ++ni) {
            const int col = n0 + wn * 64 + ni * 16 + lrow;
            const long rbase = (long)m0 + wm * TM + mi * 16 + rhalf;
#pragma unroll
            for (int r = 0; r < 8; ++r) {
                const long row = rbase + r;
                float v = acc[mi][ni][r];
                if (MODE == 0) {
                    v += aux[col];
                    v = v > 0.0f ? v : 0.0f;
                    out16[row * ldn + col] = f32_to_bf16(v);
                } else if (MODE == 1) {
                    v *= aux[row * 8 + (col >> 3)];
                    out16[row * ldn + col] = f32_to_bf16(v);
                } else {
                    out32[row * ldn + col] = v;
                }
            }
        }
    }
}

// ---------------------------------------------------------------------------
// Router tail: logits = h @ W2^T + b2, softmax over E=8  (1 thread / token)
// ---------------------------------------------------------------------------
__global__ __launch_bounds__(256) void router_kernel(
    const unsigned short* __restrict__ hb, const float* __restrict__ W2,
    const float* __restrict__ b2, const float* __restrict__ temp,
    float* __restrict__ gates) {
    __shared__ float w2s[kE][kDH];   // 16 KB, broadcast reads
    for (int i = threadIdx.x; i < kE * kDH; i += 256)
        w2s[i >> 9][i & 511] = W2[i];
    __syncthreads();

    const long t = (long)blockIdx.x * 256 + threadIdx.x;
    const unsigned short* hrow = hb + t * kDH;
    float acc[kE] = {};
    for (int k = 0; k < kDH; k += 8) {
        ushort8 hv8 = *(const ushort8*)(hrow + k);
#pragma unroll
        for (int j = 0; j < 8; ++j) {
            const float hv = bf16_to_f32(hv8[j]);
#pragma unroll
            for (int e = 0; e < kE; ++e) acc[e] += hv * w2s[e][k + j];
        }
    }
    const float invT = 1.0f / temp[0];
    float mx = -3.0e38f;
#pragma unroll
    for (int e = 0; e < kE; ++e) { acc[e] = (acc[e] + b2[e]) * invT; mx = fmaxf(mx, acc[e]); }
    float s = 0.0f;
#pragma unroll
    for (int e = 0; e < kE; ++e) { acc[e] = __expf(acc[e] - mx); s += acc[e]; }
    const float inv = 1.0f / s;
#pragma unroll
    for (int e = 0; e < kE; ++e) gates[t * kE + e] = acc[e] * inv;
}

// ---------------------------------------------------------------------------
extern "C" void kernel_launch(void* const* d_in, const int* in_sizes, int n_in,
                              void* d_out, int out_size, void* d_ws, size_t ws_size,
                              hipStream_t stream) {
    (void)in_sizes; (void)n_in; (void)out_size; (void)ws_size;
    const float* x    = (const float*)d_in[0];
    const float* Wb   = (const float*)d_in[1];
    const float* W1   = (const float*)d_in[2];
    const float* b1   = (const float*)d_in[3];
    const float* W2   = (const float*)d_in[4];
    const float* b2   = (const float*)d_in[5];
    const float* temp = (const float*)d_in[6];
    const float* lA   = (const float*)d_in[7];
    const float* lB   = (const float*)d_in[8];
    float* out = (float*)d_out;

    size_t off = 0;
    auto take = [&](size_t bytes) {
        void* p = (char*)d_ws + off;
        off += (bytes + 255) & ~(size_t)255;
        return p;
    };
    unsigned short* xb    = (unsigned short*)take((size_t)kTok * kD * 2);   // x bf16
    unsigned short* wbb   = (unsigned short*)take((size_t)kD * kD * 2);     // W_base bf16
    unsigned short* w1b   = (unsigned short*)take((size_t)kDH * kD * 2);    // W1 bf16
    unsigned short* a2b   = (unsigned short*)take((size_t)kER * kD * 2);    // lora_A [64,1024]
    unsigned short* ub    = (unsigned short*)take((size_t)kD * kER * 2);    // U [1024,64]
    unsigned short* hb    = (unsigned short*)take((size_t)kTok * kDH * 2);  // router hidden
    float*          gates = (float*)take((size_t)kTok * kE * 4);
    unsigned short* yb    = (unsigned short*)take((size_t)kTok * kER * 2);  // gated xa

    prep_kernel<<<2048, 256, 0, stream>>>(x, Wb, W1, lA, lB, xb, wbb, w1b, a2b, ub);

    // h = relu(x @ W1^T + b1)   [16384 x 512]
    gemm_bf16_kernel<0, 128><<<dim3(kTok / BM, kDH / 128), 256, 0, stream>>>(
        xb, w1b, kD, b1, nullptr, nullptr, hb, nullptr, kDH);

    // gates = softmax((h @ W2^T + b2) / T)   [16384 x 8]
    router_kernel<<<kTok / 256, 256, 0, stream>>>(hb, W2, b2, temp, gates);

    // y = (x @ A2^T) * gate_broadcast        [16384 x 64]
    gemm_bf16_kernel<1, 64><<<dim3(kTok / BM, 1), 256, 0, stream>>>(
        xb, a2b, kD, gates, nullptr, nullptr, yb, nullptr, kER);

    // out = x @ W_base^T + y @ U^T           [16384 x 1024], fp32
    gemm_bf16_kernel<2, 128><<<dim3(kTok / BM, kD / 128), 256, 0, stream>>>(
        xb, wbb, kD, nullptr, yb, ub, nullptr, out, kD);
}
